// eventfft_cross_47270410060139
// MI455X (gfx1250) — compile-verified, matrix operand order checked by reading, not run on hardware
//
#include <hip/hip_runtime.h>
#include <hip/hip_bf16.h>
#include <math.h>

// ---------------- problem constants ----------------
#define B_    4
#define C_    64
#define H_    256
#define W_    256
#define HW_   (H_ * W_)                 // 65536
#define NPIX_ ((size_t)B_ * HW_)        // 262144

typedef __bf16 bf16;
typedef __attribute__((ext_vector_type(16))) __bf16 v16bf;
typedef __attribute__((ext_vector_type(8)))  __bf16 v8bf;
typedef __attribute__((ext_vector_type(8)))  float  v8f;
typedef int v4i_vs __attribute__((vector_size(4 * sizeof(int))));

#define AS1_ __attribute__((address_space(1)))
#define AS3_ __attribute__((address_space(3)))

#if __has_builtin(__builtin_amdgcn_global_load_async_to_lds_b128) && \
    __has_builtin(__builtin_amdgcn_s_wait_asynccnt)
#define USE_ASYNC_LDS 1
#endif

// =====================================================================
// LayerNorm over channel dim of (B,Cn,HW), output bf16 (WMMA operand)
// =====================================================================
__global__ __launch_bounds__(256)
void ln_ch_to_bf16(const float* __restrict__ X, const float* __restrict__ g,
                   const float* __restrict__ bta, bf16* __restrict__ Y, int Cn) {
    size_t p = (size_t)blockIdx.x * blockDim.x + threadIdx.x;
    if (p >= NPIX_) return;
    size_t base = (p / HW_) * (size_t)Cn * HW_ + (p % HW_);
    float s = 0.f, s2 = 0.f;
    for (int c = 0; c < Cn; ++c) {
        float v = X[base + (size_t)c * HW_];
        s += v; s2 += v * v;
    }
    float mu  = s / (float)Cn;
    float var = s2 / (float)Cn - mu * mu;
    float inv = rsqrtf(var + 1e-5f);
    for (int c = 0; c < Cn; ++c) {
        float v = (X[base + (size_t)c * HW_] - mu) * inv * g[c] + bta[c];
        Y[base + (size_t)c * HW_] = (bf16)v;
    }
}

// =====================================================================
// LN over 128 channels of CORR, multiply by v (= KVD channels 128..255)
// =====================================================================
__global__ __launch_bounds__(256)
void lnA_mul_v_to_bf16(const float* __restrict__ CORR, const float* __restrict__ KVD,
                       const float* __restrict__ g, const float* __restrict__ bta,
                       bf16* __restrict__ Y) {
    size_t p = (size_t)blockIdx.x * blockDim.x + threadIdx.x;
    if (p >= NPIX_) return;
    size_t b  = p / HW_, hw = p % HW_;
    size_t baseO = b * (size_t)128 * HW_ + hw;
    size_t baseV = b * (size_t)256 * HW_ + (size_t)128 * HW_ + hw;
    float s = 0.f, s2 = 0.f;
    for (int c = 0; c < 128; ++c) {
        float v = CORR[baseO + (size_t)c * HW_];
        s += v; s2 += v * v;
    }
    float mu  = s * (1.f / 128.f);
    float var = s2 * (1.f / 128.f) - mu * mu;
    float inv = rsqrtf(var + 1e-5f);
    for (int c = 0; c < 128; ++c) {
        float o = (CORR[baseO + (size_t)c * HW_] - mu) * inv * g[c] + bta[c];
        float v = KVD[baseV + (size_t)c * HW_];
        Y[baseO + (size_t)c * HW_] = (bf16)(o * v);
    }
}

// =====================================================================
// bf16 WMMA GEMM:  D[b,o,p] = sum_c W[o,c] * X[b,c,p]  (+ optional residual)
// block = 256 threads = 8 waves; each wave owns a 16x16 output tile.
// X tile staged TRANSPOSED in LDS (pixel-major, K contiguous) so every
// lane fragment is two aligned ds_load_b128. Weight tile staged to LDS
// via global_load_async_to_lds_b128 (ASYNCcnt path).
// =====================================================================
template <int CIN>
__global__ __launch_bounds__(256)
void gemm_bf16_wmma(const bf16* __restrict__ Wm, const bf16* __restrict__ X,
                    const float* __restrict__ Rres, float* __restrict__ D, int O) {
    __shared__ __align__(16) bf16 xt[128 * CIN];  // [pixel][k]
    __shared__ __align__(16) bf16 wl[16 * CIN];   // [row][k] (row-major, as in global)
    const int b    = blockIdx.z;
    const int rowt = blockIdx.y * 16;
    const int col0 = blockIdx.x * 128;
    const int t    = threadIdx.x;
    const int wave = t >> 5;
    const int lane = t & 31;
    const int n    = lane & 15;     // col within wave tile / A row select
    const int hi   = lane >> 4;     // lane half -> K sub-block select
    const bf16* Xb = X + (size_t)b * CIN * HW_;

    // ---- stage weight tile (16 x CIN), layout identical to global ----
#if defined(USE_ASYNC_LDS)
    if (t < 2 * CIN) {
        const bf16* src = Wm + (size_t)rowt * CIN + t * 8;
        __builtin_amdgcn_global_load_async_to_lds_b128(
            (AS1_ v4i_vs*)src, (AS3_ v4i_vs*)&wl[t * 8], 0, 0);
    }
#else
    if (t < 2 * CIN)
        *(v8bf*)&wl[t * 8] = *(const v8bf*)(Wm + (size_t)rowt * CIN + t * 8);
#endif

    // ---- stage X tile transposed: xt[col][k] = X[k][col0+col] ----
#pragma unroll
    for (int e = t; e < CIN * 16; e += 256) {
        int r  = e >> 4;            // K row
        int cb = (e & 15) * 8;      // col chunk
        v8bf vv = *(const v8bf*)(Xb + (size_t)r * HW_ + col0 + cb);
#pragma unroll
        for (int i = 0; i < 8; ++i)
            xt[(cb + i) * CIN + r] = vv[i];
    }
    // prefetch the next pixel tile for this (row,batch) while we compute
    if (col0 + 128 < HW_)
        __builtin_prefetch(Xb + col0 + 128 + t, 0, 1);
#if defined(USE_ASYNC_LDS)
    __builtin_amdgcn_s_wait_asynccnt(0);
#endif
    __syncthreads();

    // ---- fully unrolled K loop: CIN/32 WMMAs ----
    v8f acc = {};
#pragma unroll
    for (int kk = 0; kk < CIN; kk += 32) {
        const int kb = kk + hi * 8;
        v8bf a0 = *(const v8bf*)&wl[n * CIN + kb];
        v8bf a1 = *(const v8bf*)&wl[n * CIN + kb + 16];
        v8bf b0 = *(const v8bf*)&xt[(wave * 16 + n) * CIN + kb];
        v8bf b1 = *(const v8bf*)&xt[(wave * 16 + n) * CIN + kb + 16];
        v16bf af = __builtin_shufflevector(a0, a1, 0, 1, 2, 3, 4, 5, 6, 7,
                                           8, 9, 10, 11, 12, 13, 14, 15);
        v16bf bf = __builtin_shufflevector(b0, b1, 0, 1, 2, 3, 4, 5, 6, 7,
                                           8, 9, 10, 11, 12, 13, 14, 15);
        acc = __builtin_amdgcn_wmma_f32_16x16x32_bf16(
            false, af, false, bf, (short)0, acc, false, false);
    }

    // ---- epilogue: D elem r -> row rowt + r + hi*8, col col0+wave*16+n ----
    float* Db       = D + (size_t)b * O * HW_;
    const float* Rb = Rres ? Rres + (size_t)b * O * HW_ : nullptr;
#pragma unroll
    for (int r = 0; r < 8; ++r) {
        int row    = rowt + r + hi * 8;
        size_t idx = (size_t)row * HW_ + col0 + wave * 16 + n;
        float val  = acc[r];
        if (Rb) val += Rb[idx];
        Db[idx] = val;
    }
}

// =====================================================================
// Depthwise 3x3, SAME padding, layout (B,Cn,H,W)
// =====================================================================
__global__ __launch_bounds__(256)
void dwconv3(const float* __restrict__ X, const float* __restrict__ Wd,
             float* __restrict__ Y, int Cn) {
    size_t i = (size_t)blockIdx.x * blockDim.x + threadIdx.x;
    size_t total = (size_t)B_ * Cn * HW_;
    if (i >= total) return;
    int hw   = (int)(i % HW_);
    size_t bc = i / HW_;
    int c    = (int)(bc % Cn);
    int x0   = hw & (W_ - 1);
    int y0   = hw >> 8;
    const float* w  = Wd + c * 9;
    const float* Xp = X + bc * HW_;
    float s = 0.f;
#pragma unroll
    for (int dy = -1; dy <= 1; ++dy)
#pragma unroll
        for (int dx = -1; dx <= 1; ++dx) {
            int yy = y0 + dy, xx = x0 + dx;
            if (yy >= 0 && yy < H_ && xx >= 0 && xx < W_)
                s += Xp[yy * W_ + xx] * w[(dy + 1) * 3 + (dx + 1)];
        }
    Y[i] = s;
}

// =====================================================================
// Attention: irfft2(rfft2(q)*rfft2(k)) == 8x8 circular convolution.
// =====================================================================
__global__ __launch_bounds__(256)
void patch_ccorr(const float* __restrict__ Q, const float* __restrict__ KVD,
                 float* __restrict__ Oc) {
    __shared__ float sq[4][64];
    __shared__ float sk[4][64];
    int g  = threadIdx.x >> 6;
    int tl = threadIdx.x & 63;
    size_t unit = (size_t)blockIdx.x * 4 + g;
    int px = (int)(unit & 31);
    int py = (int)((unit >> 5) & 31);
    size_t bc = unit >> 10;           // b*128 + ch
    size_t b  = bc >> 7;
    size_t ch = bc & 127;
    size_t off   = (size_t)(py * 8) * W_ + px * 8;
    size_t qbase = bc * HW_ + off;
    size_t kbase = (b * 256 + ch) * HW_ + off;
    int li = tl >> 3, lj = tl & 7;
    sq[g][tl] = Q[qbase + li * W_ + lj];
    sk[g][tl] = KVD[kbase + li * W_ + lj];
    __syncthreads();
    int m = tl >> 3, n = tl & 7;
    float s = 0.f;
#pragma unroll
    for (int i = 0; i < 8; ++i)
#pragma unroll
        for (int j = 0; j < 8; ++j)
            s += sq[g][i * 8 + j] * sk[g][((m - i) & 7) * 8 + ((n - j) & 7)];
    Oc[qbase + m * W_ + n] = s;
}

// =====================================================================
// DFFN spectral filter: h[c] = irfft2(fft_param[c]) via Hermitian-extended
// inverse DFT of the real 8x5 half-spectrum.
// =====================================================================
__global__ __launch_bounds__(64)
void build_hker(const float* __restrict__ Pm, float* __restrict__ Hk) {
    int c = blockIdx.x;
    int t = threadIdx.x;
    int m = t >> 3, n = t & 7;
    float s = 0.f;
    for (int u = 0; u < 8; ++u)
        for (int v = 0; v < 8; ++v) {
            float Mv = (v <= 4) ? Pm[c * 40 + u * 5 + v]
                                : Pm[c * 40 + ((8 - u) & 7) * 5 + (8 - v)];
            int ph = (u * m + v * n) & 7;
            s += Mv * cosf(0.78539816339f * (float)ph); // 2*pi*ph/8
        }
    Hk[c * 64 + t] = s * (1.f / 64.f);
}

// =====================================================================
// DFFN: y (*) h[c] per 8x8 patch (== irfft2(rfft2(y)*fft_param))
// =====================================================================
__global__ __launch_bounds__(256)
void patch_filter(const float* __restrict__ Yin, const float* __restrict__ Hk,
                  float* __restrict__ Oc) {
    __shared__ float sy[4][64];
    __shared__ float sh[4][64];
    int g  = threadIdx.x >> 6;
    int tl = threadIdx.x & 63;
    size_t unit = (size_t)blockIdx.x * 4 + g;
    int px = (int)(unit & 31);
    int py = (int)((unit >> 5) & 31);
    size_t bc = unit >> 10;           // b*256 + ch
    size_t ch = bc & 255;
    size_t base = bc * HW_ + (size_t)(py * 8) * W_ + px * 8;
    int li = tl >> 3, lj = tl & 7;
    sy[g][tl] = Yin[base + li * W_ + lj];
    sh[g][tl] = Hk[ch * 64 + tl];
    __syncthreads();
    int m = tl >> 3, n = tl & 7;
    float s = 0.f;
#pragma unroll
    for (int i = 0; i < 8; ++i)
#pragma unroll
        for (int j = 0; j < 8; ++j)
            s += sy[g][i * 8 + j] * sh[g][((m - i) & 7) * 8 + ((n - j) & 7)];
    Oc[base + m * W_ + n] = s;
}

// =====================================================================
// exact-erf GELU gate: G[b,c,hw] = gelu(Yd[b,c]) * Yd[b,c+128]  -> bf16
// =====================================================================
__global__ __launch_bounds__(256)
void gelu_gate_to_bf16(const float* __restrict__ Yd, bf16* __restrict__ G) {
    size_t i = (size_t)blockIdx.x * blockDim.x + threadIdx.x;
    size_t total = (size_t)B_ * 128 * HW_;
    if (i >= total) return;
    size_t hw = i % HW_;
    size_t bc = i / HW_;
    size_t b  = bc >> 7;
    size_t c  = bc & 127;
    float a  = Yd[(b * 256 + c) * HW_ + hw];
    float g2 = Yd[(b * 256 + 128 + c) * HW_ + hw];
    float ge = 0.5f * a * (1.f + erff(a * 0.70710678118f));
    G[i] = (bf16)(ge * g2);
}

// =====================================================================
// f32 -> bf16 weight convert
// =====================================================================
__global__ __launch_bounds__(256)
void cvt_bf16(const float* __restrict__ src, bf16* __restrict__ dst, int n) {
    int i = blockIdx.x * blockDim.x + threadIdx.x;
    if (i < n) dst[i] = (bf16)src[i];
}

// ---------------------------------------------------------------------
extern "C" void kernel_launch(void* const* d_in, const int* in_sizes, int n_in,
                              void* d_out, int out_size, void* d_ws, size_t ws_size,
                              hipStream_t stream) {
    const float* x      = (const float*)d_in[0];
    const float* evt    = (const float*)d_in[1];
    const float* ln1i_w = (const float*)d_in[2];
    const float* ln1i_b = (const float*)d_in[3];
    const float* ln1e_w = (const float*)d_in[4];
    const float* ln1e_b = (const float*)d_in[5];
    const float* w_q    = (const float*)d_in[6];
    const float* dw_q   = (const float*)d_in[7];
    const float* w_kv   = (const float*)d_in[8];
    const float* dw_kv  = (const float*)d_in[9];
    const float* lnA_w  = (const float*)d_in[10];
    const float* lnA_b  = (const float*)d_in[11];
    const float* w_poA  = (const float*)d_in[12];
    const float* ln2_w  = (const float*)d_in[13];
    const float* ln2_b  = (const float*)d_in[14];
    const float* w_pi   = (const float*)d_in[15];
    const float* fftp   = (const float*)d_in[16];
    const float* dw_f   = (const float*)d_in[17];
    const float* w_poF  = (const float*)d_in[18];

    const size_t MB = 1ull << 20;
    char* ws = (char*)d_ws;
    bf16*  XNBF = (bf16*) (ws + 0);          // 32MB: ln1(x) bf16, later ln2(xnew)
    bf16*  ENBF = (bf16*) (ws + 32  * MB);   // 32MB: ln1(evt) bf16
    bf16*  WBF  = (bf16*) (ws + 64  * MB);   // packed bf16 weights
    float* HKER = (float*)(ws + 65  * MB);   // 256x64 spectral->spatial kernels
    float* XNEW = (float*)(ws + 66  * MB);   // 64MB: x after attention residual
    bf16*  VOUT = (bf16*) (ws + 130 * MB);   // 64MB: (v*lnA) bf16, later gate bf16
    float* Qb   = (float*)(ws + 194 * MB);   // 128MB: q, later corr
    float* QD   = (float*)(ws + 322 * MB);   // 128MB: dwconv(q)
    float* KV   = (float*)(ws + 450 * MB);   // 256MB: kv, later y, later yd
    float* KVD  = (float*)(ws + 706 * MB);   // 256MB: dwconv(kv), later yc

    bf16* WQ  = WBF + 0;          // 128x64
    bf16* WKV = WBF + 8192;       // 256x64
    bf16* WPA = WBF + 24576;      // 64x128
    bf16* WPI = WBF + 32768;      // 256x64
    bf16* WPF = WBF + 49152;      // 64x128

    // weight converts + spectral kernels
    cvt_bf16<<<32, 256, 0, stream>>>(w_q,   WQ,  8192);
    cvt_bf16<<<64, 256, 0, stream>>>(w_kv,  WKV, 16384);
    cvt_bf16<<<32, 256, 0, stream>>>(w_poA, WPA, 8192);
    cvt_bf16<<<64, 256, 0, stream>>>(w_pi,  WPI, 16384);
    cvt_bf16<<<32, 256, 0, stream>>>(w_poF, WPF, 8192);
    build_hker<<<256, 64, 0, stream>>>(fftp, HKER);

    const int LNB = (int)(NPIX_ / 256);

    // ---- attention branch ----
    ln_ch_to_bf16<<<LNB, 256, 0, stream>>>(x,   ln1i_w, ln1i_b, XNBF, C_);
    ln_ch_to_bf16<<<LNB, 256, 0, stream>>>(evt, ln1e_w, ln1e_b, ENBF, C_);

    gemm_bf16_wmma<64><<<dim3(HW_ / 128, 8,  B_), 256, 0, stream>>>(WQ,  XNBF, nullptr, Qb, 128);
    gemm_bf16_wmma<64><<<dim3(HW_ / 128, 16, B_), 256, 0, stream>>>(WKV, ENBF, nullptr, KV, 256);

    dwconv3<<<(int)((size_t)B_ * 128 * HW_ / 256), 256, 0, stream>>>(Qb, dw_q,  QD,  128);
    dwconv3<<<(int)((size_t)B_ * 256 * HW_ / 256), 256, 0, stream>>>(KV, dw_kv, KVD, 256);

    // corr = irfft2(rfft2(q)*rfft2(k)) written into Qb (unpatchified layout)
    patch_ccorr<<<(int)((size_t)B_ * 128 * 1024 / 4), 256, 0, stream>>>(QD, KVD, Qb);

    lnA_mul_v_to_bf16<<<LNB, 256, 0, stream>>>(Qb, KVD, lnA_w, lnA_b, VOUT);

    // x_new = x + conv1x1(v*out, w_po_attn)
    gemm_bf16_wmma<128><<<dim3(HW_ / 128, 4, B_), 256, 0, stream>>>(WPA, VOUT, x, XNEW, 64);

    // ---- DFFN branch ----
    ln_ch_to_bf16<<<LNB, 256, 0, stream>>>(XNEW, ln2_w, ln2_b, XNBF, C_);
    gemm_bf16_wmma<64><<<dim3(HW_ / 128, 16, B_), 256, 0, stream>>>(WPI, XNBF, nullptr, KV, 256);

    // yc = irfft2(rfft2(y)*fft_param)  (per-channel circular filter)
    patch_filter<<<(int)((size_t)B_ * 256 * 1024 / 4), 256, 0, stream>>>(KV, HKER, KVD);

    // yd = dwconv3x3(yc)
    dwconv3<<<(int)((size_t)B_ * 256 * HW_ / 256), 256, 0, stream>>>(KVD, dw_f, KV, 256);

    // gate = gelu(y1) * y2  -> bf16
    gelu_gate_to_bf16<<<(int)((size_t)B_ * 128 * HW_ / 256), 256, 0, stream>>>(KV, VOUT);

    // out = x_new + conv1x1(gate, w_po_ffn)
    gemm_bf16_wmma<128><<<dim3(HW_ / 128, 4, B_), 256, 0, stream>>>(WPF, VOUT, XNEW, (float*)d_out, 64);
}
